// DualHazardHead_64793876628258
// MI455X (gfx1250) — matrix operand order repaired and checked by reading, not compiled, not run on hardware
//
#include <hip/hip_runtime.h>
#include <hip/hip_bf16.h>
#include <stdint.h>

// ---------------- types ----------------
typedef __attribute__((ext_vector_type(16))) __bf16 v16bf;
typedef __attribute__((ext_vector_type(8)))  float  v8f;

union FragBF { v16bf v; unsigned u[8]; };

__device__ __forceinline__ unsigned bf16pack(float a, float b) {
  unsigned ua = __float_as_uint(a); ua = (ua + 0x7FFFu + ((ua >> 16) & 1u)) >> 16;
  unsigned ub = __float_as_uint(b); ub = (ub + 0x7FFFu + ((ub >> 16) & 1u)) >> 16;
  return ua | (ub << 16);
}
__device__ __forceinline__ unsigned short bf16one(float a) {
  unsigned ua = __float_as_uint(a);
  return (unsigned short)((ua + 0x7FFFu + ((ua >> 16) & 1u)) >> 16);
}

// ---------------- problem constants ----------------
#define NTOK   65536          // B*T
#define MT     64             // tokens per block
#define NT1    272            // trunk out cols: 256 z + 7 router + 9 pad (17 tiles)
#define NHEAD  320            // 20 inst_base + 140 inst_exp + 20 grp_base + 140 grp_exp
#define KX_U32 304            // padded trunk K (608 bf16) in dwords (global pitch)
#define APITCH 308            // LDS A row pitch in u32 (1232 B, 16B aligned, conflict-free)
#define ZPITCH 132            // LDS z row pitch in u32 (528 B, 16B aligned)
#define EPITCH 321            // LDS e row pitch in f32 (odd)
#define BPITCH 20             // LDS B-slice row pitch in u32 (80 B, 16B aligned)

// LDS layout (bytes)
#define OFF_Z  82176          // e region [64][321] f32 = 82176 B (A overlays: [64][308] u32 = 78848)
#define OFF_B  115968         // z region [64][132] u32 = 33792 B
#define OFF_PW 141568         // B slice [320][20] u32 = 25600 B
#define SMEM_TOTAL 143616     // + pw [64][8] f32 = 2048 B

// ws layout (bytes)
#define WS_WBIG 0
#define WS_WH   330752        // wbig = 272*304*4
#define WS_BB   494592        // wh   = 320*128*4
#define WS_BH   495680        // bb   = 272*4 ; bh = 320*4

// ---------------- CDNA5 async global->LDS copy (16 B per lane) ----------------
__device__ __forceinline__ void async_g2l_b128(unsigned ldsAddr, const void* gaddr) {
  asm volatile("global_load_async_to_lds_b128 %0, %1, off"
               :: "v"(ldsAddr), "v"((unsigned long long)(uintptr_t)gaddr)
               : "memory");
}
__device__ __forceinline__ void wait_async_zero() {
  asm volatile("s_wait_asynccnt 0x0" ::: "memory");
}

// ---------------- fragment loaders (2 x ds_load_b128 each) ----------------
__device__ __forceinline__ void loadFrag(FragBF& f, const unsigned* base) {
  uint4 a = *(const uint4*)base;
  uint4 b = *(const uint4*)(base + 8);   // +32 B
  f.u[0] = a.x; f.u[1] = a.y; f.u[2] = a.z; f.u[3] = a.w;
  f.u[4] = b.x; f.u[5] = b.y; f.u[6] = b.z; f.u[7] = b.w;
}
__device__ __forceinline__ void loadB(FragBF& f, const unsigned* Bb, int n, int khalf) {
  const unsigned* bp = Bb + n * BPITCH + khalf * 4;
  uint4 a = *(const uint4*)bp;
  uint4 b = *(const uint4*)(bp + 8);     // +32 B
  f.u[0] = a.x; f.u[1] = a.y; f.u[2] = a.z; f.u[3] = a.w;
  f.u[4] = b.x; f.u[5] = b.y; f.u[6] = b.z; f.u[7] = b.w;
}

// ================= prep: pack weights to bf16, [N][K] K-contiguous =================
__global__ void dhh_prep(const float* __restrict__ trunk_w, const float* __restrict__ trunk_b,
                         const float* __restrict__ inst_base_w, const float* __restrict__ inst_base_b,
                         const float* __restrict__ group_base_w, const float* __restrict__ group_base_b,
                         const float* __restrict__ inst_exp_w, const float* __restrict__ inst_exp_b,
                         const float* __restrict__ group_exp_w, const float* __restrict__ group_exp_b,
                         const float* __restrict__ router_w, const float* __restrict__ router_b,
                         unsigned* __restrict__ wbig, unsigned* __restrict__ wh,
                         float* __restrict__ bb, float* __restrict__ bh) {
  int idx = blockIdx.x * 256 + threadIdx.x;
  if (idx < NT1 * KX_U32) {                       // trunk (+router) matrix, K padded 594->608
    int n = idx / KX_U32, kp = idx % KX_U32, k = kp * 2;
    float f0 = 0.f, f1 = 0.f;
    if (n < 256) {
      if (k     < 594) f0 = trunk_w[k * 256 + n];
      if (k + 1 < 594) f1 = trunk_w[(k + 1) * 256 + n];
    } else if (n < 263) {
      int c = n - 256;
      if (k     < 512) f0 = router_w[k * 7 + c];
      if (k + 1 < 512) f1 = router_w[(k + 1) * 7 + c];
    }
    wbig[n * KX_U32 + kp] = bf16pack(f0, f1);
    return;
  }
  int i2 = idx - NT1 * KX_U32;
  if (i2 >= 0 && i2 < NHEAD * 128) {              // head matrix [320][256]
    int n = i2 / 128, kp = i2 % 128, k = kp * 2;
    float f0, f1;
    if (n < 20)        { f0 = inst_base_w[k * 20 + n];            f1 = inst_base_w[(k + 1) * 20 + n]; }
    else if (n < 160)  { int p = (n - 20) / 20, c = (n - 20) % 20;
                         const float* w = inst_exp_w + p * 5120;  f0 = w[k * 20 + c]; f1 = w[(k + 1) * 20 + c]; }
    else if (n < 180)  { int c = n - 160; f0 = group_base_w[k * 20 + c]; f1 = group_base_w[(k + 1) * 20 + c]; }
    else               { int p = (n - 180) / 20, c = (n - 180) % 20;
                         const float* w = group_exp_w + p * 5120; f0 = w[k * 20 + c]; f1 = w[(k + 1) * 20 + c]; }
    wh[n * 128 + kp] = bf16pack(f0, f1);
    return;
  }
  int i3 = i2 - NHEAD * 128;
  if (i3 >= 0 && i3 < NT1) {
    float v = 0.f;
    if (i3 < 256) v = trunk_b[i3]; else if (i3 < 263) v = router_b[i3 - 256];
    bb[i3] = v;
    return;
  }
  int i4 = i3 - NT1;
  if (i4 >= 0 && i4 < NHEAD) {
    float v;
    if (i4 < 20)       v = inst_base_b[i4];
    else if (i4 < 160) v = inst_exp_b[i4 - 20];
    else if (i4 < 180) v = group_base_b[i4 - 160];
    else               v = group_exp_b[i4 - 180];
    bh[i4] = v;
  }
}

// ================= fused trunk + router + heads + combine =================
__global__ void dhh_fused(const float* __restrict__ h_t, const float* __restrict__ a_t,
                          const float* __restrict__ d_t, const float* __restrict__ age,
                          const unsigned* __restrict__ wbig, const unsigned* __restrict__ wh,
                          const float* __restrict__ bb, const float* __restrict__ bh,
                          float* __restrict__ out_inst, float* __restrict__ out_group) {
  extern __shared__ char smem[];
  unsigned*       A  = (unsigned*)smem;                 // [64][308] u32  (bf16 pairs)
  float*          E  = (float*)smem;                    // [64][321] f32  (overlays A, used later)
  unsigned short* Zh = (unsigned short*)(smem + OFF_Z); // [64][264] bf16
  unsigned*       Zu = (unsigned*)(smem + OFF_Z);       // same, dword view, pitch 132
  unsigned*       Bb = (unsigned*)(smem + OFF_B);       // [320][20] u32 slice
  float*          PW = (float*)(smem + OFF_PW);         // [64][8] f32

  const int tid   = threadIdx.x;
  const int lane  = tid & 31;
  const int wave  = tid >> 5;
  const int mg    = wave >> 1;      // 0..3  -> rows [mg*16, +16)
  const int nh    = wave & 1;       // 0..1  -> column half
  const int nloc  = lane & 15;
  const int khalf = lane >> 4;
  const int tok0  = blockIdx.x * MT;
  const unsigned ldsB = (unsigned)(uintptr_t)(smem + OFF_B);  // LDS byte addr of B slice

  // ---- stage A tile: x = concat(h,a,d,age) fp32 -> bf16, zero-pad K to 608+ ----
  for (int i = tid; i < MT * APITCH; i += 256) {
    int m = i / APITCH, kp = i % APITCH, k = kp * 2;
    int tok = tok0 + m;
    float f0 = 0.f, f1 = 0.f;
    if (k < 594) {  // segment boundaries are even -> pair never straddles
      if (k < 512)      { const float* p = h_t + tok * 512 + k;         f0 = p[0]; f1 = p[1]; }
      else if (k < 576) { const float* p = a_t + tok * 64  + (k - 512); f0 = p[0]; f1 = p[1]; }
      else if (k < 578) { const float* p = d_t + tok * 2   + (k - 576); f0 = p[0]; f1 = p[1]; }
      else              { const float* p = age + tok * 16  + (k - 578); f0 = p[0]; f1 = p[1]; }
    }
    A[m * APITCH + kp] = bf16pack(f0, f1);
  }

  // ---- trunk GEMM: [64 x 608] * [608 x 272] ----
  const int ntStart = nh ? 9 : 0;
  const int ntCount = nh ? 8 : 9;
  v8f acc[9] = {};
  for (int kc = 0; kc < 19; ++kc) {
    __syncthreads();
    // stage W slice [272][32] bf16 via async global->LDS (16 B per lane)
    for (int i = tid; i < NT1 * 4; i += 256) {
      int n = i >> 2, j = i & 3;
      async_g2l_b128(ldsB + n * (BPITCH * 4) + j * 16,
                     (const char*)wbig + (size_t)n * (KX_U32 * 4) + kc * 64 + j * 16);
    }
    wait_async_zero();
    __syncthreads();
    FragBF af;
    loadFrag(af, A + (mg * 16 + nloc) * APITCH + kc * 16 + khalf * 4);
    FragBF bf0, bf1;
    loadB(bf0, Bb, ntStart * 16 + nloc, khalf);
    #pragma unroll
    for (int t = 0; t < 9; ++t) {
      if (t < ntCount) {                          // wave-uniform guard, EXEC stays full
        FragBF& cur = (t & 1) ? bf1 : bf0;
        FragBF& nxt = (t & 1) ? bf0 : bf1;
        if (t + 1 < ntCount)
          loadB(nxt, Bb, (ntStart + t + 1) * 16 + nloc, khalf);
        acc[t] = __builtin_amdgcn_wmma_f32_16x16x32_bf16(
            false, af.v, false, cur.v, (short)0, acc[t], false, false);
      }
    }
  }

  // ---- trunk epilogue: GELU -> z (bf16), raw router logits -> PW ----
  #pragma unroll
  for (int t = 0; t < 9; ++t) {
    if (t < ntCount) {
      int nt = ntStart + t;
      int n  = nt * 16 + nloc;
      float bias = bb[n];
      #pragma unroll
      for (int v = 0; v < 8; ++v) {
        int row = mg * 16 + khalf * 8 + v;
        float val = acc[t][v] + bias;
        if (nt < 16) {
          float g = 0.5f * val * (1.0f + erff(val * 0.70710678118654752f));
          Zh[row * (ZPITCH * 2) + n] = bf16one(g);
        } else if (nloc < 7) {
          PW[row * 8 + nloc] = val;
        }
      }
    }
  }
  __syncthreads();

  // ---- softmax over 7 phases ----
  if (tid < MT) {
    float l[7], mx = -1e30f;
    #pragma unroll
    for (int p = 0; p < 7; ++p) { l[p] = PW[tid * 8 + p]; mx = fmaxf(mx, l[p]); }
    float s = 0.f;
    #pragma unroll
    for (int p = 0; p < 7; ++p) { l[p] = __expf(l[p] - mx); s += l[p]; }
    float inv = 1.0f / s;
    #pragma unroll
    for (int p = 0; p < 7; ++p) PW[tid * 8 + p] = l[p] * inv;
  }
  __syncthreads();

  // ---- heads GEMM: [64 x 256] * [256 x 320] ----
  v8f hacc[10] = {};
  for (int kc = 0; kc < 8; ++kc) {
    __syncthreads();
    for (int i = tid; i < NHEAD * 4; i += 256) {  // stage head-W slice [320][32] bf16
      int n = i >> 2, j = i & 3;
      async_g2l_b128(ldsB + n * (BPITCH * 4) + j * 16,
                     (const char*)wh + (size_t)n * 512 + kc * 64 + j * 16);
    }
    wait_async_zero();
    __syncthreads();
    FragBF af;
    loadFrag(af, Zu + (mg * 16 + nloc) * ZPITCH + kc * 16 + khalf * 4);
    FragBF bf0, bf1;
    loadB(bf0, Bb, (nh * 10) * 16 + nloc, khalf);
    #pragma unroll
    for (int t = 0; t < 10; ++t) {
      FragBF& cur = (t & 1) ? bf1 : bf0;
      FragBF& nxt = (t & 1) ? bf0 : bf1;
      if (t + 1 < 10)
        loadB(nxt, Bb, (nh * 10 + t + 1) * 16 + nloc, khalf);
      hacc[t] = __builtin_amdgcn_wmma_f32_16x16x32_bf16(
          false, af.v, false, cur.v, (short)0, hacc[t], false, false);
    }
  }

  // ---- heads epilogue: e = acc + bias -> LDS (overlays A region; trunk done) ----
  #pragma unroll
  for (int t = 0; t < 10; ++t) {
    int n = (nh * 10 + t) * 16 + nloc;
    float bias = bh[n];
    #pragma unroll
    for (int v = 0; v < 8; ++v) {
      int row = mg * 16 + khalf * 8 + v;
      E[row * EPITCH + n] = hacc[t][v] + bias;
    }
  }
  __syncthreads();

  // ---- combine: out = base + sum_p pw[p] * expert_p ----
  for (int i = tid; i < MT * 20; i += 256) {
    int m = i / 20, k = i % 20;
    const float* er = E  + m * EPITCH;
    const float* pw = PW + m * 8;
    float vi = er[k];
    float vg = er[160 + k];
    #pragma unroll
    for (int p = 0; p < 7; ++p) {
      float w = pw[p];
      vi += w * er[20  + p * 20 + k];
      vg += w * er[180 + p * 20 + k];
    }
    int tok = tok0 + m;
    out_inst[tok * 20 + k]  = vi;
    out_group[tok * 20 + k] = vg;
  }
}

// ================= host =================
extern "C" void kernel_launch(void* const* d_in, const int* in_sizes, int n_in,
                              void* d_out, int out_size, void* d_ws, size_t ws_size,
                              hipStream_t stream) {
  (void)in_sizes; (void)n_in; (void)out_size; (void)ws_size;
  const float* h_t          = (const float*)d_in[0];
  const float* a_t          = (const float*)d_in[1];
  const float* d_t          = (const float*)d_in[2];
  const float* age_embed    = (const float*)d_in[3];
  const float* trunk_w      = (const float*)d_in[4];
  const float* trunk_b      = (const float*)d_in[5];
  const float* inst_base_w  = (const float*)d_in[6];
  const float* inst_base_b  = (const float*)d_in[7];
  const float* group_base_w = (const float*)d_in[8];
  const float* group_base_b = (const float*)d_in[9];
  const float* inst_exp_w   = (const float*)d_in[10];
  const float* inst_exp_b   = (const float*)d_in[11];
  const float* group_exp_w  = (const float*)d_in[12];
  const float* group_exp_b  = (const float*)d_in[13];
  const float* router_w     = (const float*)d_in[14];
  const float* router_b     = (const float*)d_in[15];

  unsigned char* ws = (unsigned char*)d_ws;
  unsigned* wbig = (unsigned*)(ws + WS_WBIG);
  unsigned* wh   = (unsigned*)(ws + WS_WH);
  float*    bb   = (float*)(ws + WS_BB);
  float*    bh   = (float*)(ws + WS_BH);

  int prep_items = NT1 * KX_U32 + NHEAD * 128 + NT1 + NHEAD;
  dhh_prep<<<(prep_items + 255) / 256, 256, 0, stream>>>(
      trunk_w, trunk_b, inst_base_w, inst_base_b, group_base_w, group_base_b,
      inst_exp_w, inst_exp_b, group_exp_w, group_exp_b, router_w, router_b,
      wbig, wh, bb, bh);

  float* out_inst  = (float*)d_out;
  float* out_group = out_inst + (size_t)NTOK * 20;
  dhh_fused<<<NTOK / MT, 256, SMEM_TOTAL, stream>>>(
      h_t, a_t, d_t, age_embed, wbig, wh, bb, bh, out_inst, out_group);
}
